// Cowan_network_29789893165160
// MI455X (gfx1250) — matrix operand order repaired and checked by reading, not compiled, not run on hardware
//
#include <hip/hip_runtime.h>
#include <hip/hip_bf16.h>
#include <math.h>

typedef __attribute__((ext_vector_type(16))) _Float16 v16h;
typedef __attribute__((ext_vector_type(8)))  _Float16 v8h;
typedef __attribute__((ext_vector_type(8)))  float    v8f;
typedef __attribute__((ext_vector_type(4)))  unsigned v4u;
typedef __attribute__((ext_vector_type(8)))  int      v8i_;
typedef __attribute__((ext_vector_type(4)))  int      v4i_;

// explicit LDS (addrspace 3) access type — bypasses address-space inference so
// volatile 128-bit reads select directly to ds_load_b128
typedef __attribute__((address_space(3))) v8h as3_v8h;

#define WMMA_F16(a, b, c) \
  __builtin_amdgcn_wmma_f32_16x16x32_f16(false, (a), false, (b), (short)0, (c), false, false)

#if defined(__has_builtin)
#  if __has_builtin(__builtin_amdgcn_tensor_load_to_lds)
#    define HAS_TDM 1
#  endif
#endif
#ifndef HAS_TDM
#  define HAS_TDM 0
#endif

// volatile: the TDM writes LDS behind the compiler's back (descriptor-based);
// non-volatile loads from an apparently never-written LDS object get folded.
__device__ inline v8h lds_read_v8h(unsigned byteOff) {
  return *(const volatile as3_v8h*)byteOff;
}

// ---------------------------------------------------------------------------
// fragment load helpers (contiguous 8-half runs, 16B aligned)
// ---------------------------------------------------------------------------
__device__ inline v8h load8g(const _Float16* __restrict__ p, int k, int K) {
  if (k + 8 <= K) return *(const v8h*)(p + k);
  v8h z = {};
  return z; // K is always a multiple of 8 in this pipeline
}
__device__ inline v16h cat16(v8h lo, v8h hi) {
  return __builtin_shufflevector(lo, hi, 0, 1, 2, 3, 4, 5, 6, 7,
                                 8, 9, 10, 11, 12, 13, 14, 15);
}
// A fragment: 16-bit 16x32, lane group g holds K = g*16+e, contiguous
__device__ inline v16h mkA(const _Float16* __restrict__ row, int k0, int g, int K) {
  return cat16(load8g(row, k0 + g * 16, K), load8g(row, k0 + g * 16 + 8, K));
}
// B fragment: lane group g holds K = g*16+e, contiguous per column row
__device__ inline v16h mkB(const _Float16* __restrict__ row, int k0, int g, int K) {
  return cat16(load8g(row, k0 + g * 16, K), load8g(row, k0 + g * 16 + 8, K));
}

// ---------------------------------------------------------------------------
// TDM: 2D tile (tile_d1 rows x tile_d0 elems, 2-byte elements) -> LDS
// D# packed per CDNA5 ISA ch.8 (group0 / group1 bit layout)
// ---------------------------------------------------------------------------
#if HAS_TDM
__device__ inline void tdm_load_2d(unsigned lds_off, const _Float16* gaddr,
                                   unsigned tensor_d0, unsigned tensor_d1,
                                   unsigned tile_d0, unsigned tile_d1,
                                   unsigned stride0_elems) {
  unsigned long long ga = (unsigned long long)(size_t)gaddr;
  v4u g0;
  g0.x = 1u;                                            // count=1, no gather
  g0.y = lds_off;                                       // lds_addr (bytes)
  g0.z = (unsigned)(ga & 0xFFFFFFFFu);                  // global_addr[31:0]
  g0.w = (unsigned)((ga >> 32) & 0x1FFFFFFu) | (2u << 30); // addr[56:32] | type=2
  v8i_ g1;
  g1[0] = (int)(1u << 16);                              // data_size=1 (2 bytes)
  g1[1] = (int)((tensor_d0 & 0xFFFFu) << 16);           // tensor_dim0[15:0]
  g1[2] = (int)((tensor_d0 >> 16) | ((tensor_d1 & 0xFFFFu) << 16));
  g1[3] = (int)((tensor_d1 >> 16) | (tile_d0 << 16));   // tile_dim0
  g1[4] = (int)tile_d1;                                 // tile_dim1, tile_dim2=0
  g1[5] = (int)stride0_elems;                           // dim0 stride [31:0]
  g1[6] = 0;                                            // stride hi / dim1 stride
  g1[7] = 0;
  v4i_ gz = {0, 0, 0, 0};
#if defined(__clang_major__) && __clang_major__ >= 23
  v8i_ gz8 = {0, 0, 0, 0, 0, 0, 0, 0};
  __builtin_amdgcn_tensor_load_to_lds(g0, g1, gz, gz, gz8, 0);
#else
  __builtin_amdgcn_tensor_load_to_lds(g0, g1, gz, gz, 0);
#endif
}
#endif

// ---------------------------------------------------------------------------
// elementwise helpers
// ---------------------------------------------------------------------------
__global__ void k_f32_to_f16(const float* __restrict__ in, _Float16* __restrict__ out, size_t n) {
  size_t i = (size_t)blockIdx.x * blockDim.x + threadIdx.x;
  if (i < n) out[i] = (_Float16)in[i];
}

// conv weight transform: [ky][kx][ci][co] f32 -> [co][(ky*3+kx)*Cin+ci] f16
__global__ void k_wconv_t(const float* __restrict__ w, _Float16* __restrict__ wt,
                          int Cin, int Cout) {
  size_t idx = (size_t)blockIdx.x * blockDim.x + threadIdx.x;
  size_t n = (size_t)9 * Cin * Cout;
  if (idx >= n) return;
  int co = (int)(idx % Cout);
  size_t k = idx / Cout;
  wt[(size_t)co * (9 * Cin) + k] = (_Float16)w[idx];
}

__global__ void k_maxpool(const _Float16* __restrict__ in, _Float16* __restrict__ out,
                          int NB, int H, int W, int C) {
  int OH = H >> 1, OW = W >> 1;
  size_t n = (size_t)NB * OH * OW * C;
  size_t idx = (size_t)blockIdx.x * blockDim.x + threadIdx.x;
  if (idx >= n) return;
  int c = (int)(idx % C);
  size_t t = idx / C;
  int ox = (int)(t % OW); t /= OW;
  int oy = (int)(t % OH);
  int b  = (int)(t / OH);
  size_t base = (((size_t)b * H + oy * 2) * W + ox * 2) * C + c;
  float a0 = (float)in[base];
  float a1 = (float)in[base + C];
  float a2 = (float)in[base + (size_t)W * C];
  float a3 = (float)in[base + (size_t)W * C + C];
  float m = a0 > a1 ? a0 : a1;
  float m2 = a2 > a3 ? a2 : a3;
  out[idx] = (_Float16)(m > m2 ? m : m2);
}

__global__ void k_bn(const _Float16* __restrict__ x, const float* __restrict__ g,
                     const float* __restrict__ bb, const float* __restrict__ m,
                     const float* __restrict__ v, _Float16* __restrict__ y, int K) {
  size_t idx = (size_t)blockIdx.x * blockDim.x + threadIdx.x;
  size_t n = (size_t)8 * K;
  if (idx >= n) return;
  int f = (int)(idx % K);
  float s  = g[f] * rsqrtf(v[f] + 0.001f);
  float sh = bb[f] - m[f] * s;
  y[idx] = (_Float16)(s * (float)x[idx] + sh);
}

__global__ void k_buildQ(const float* __restrict__ btr, const float* __restrict__ bfix,
                         const float* __restrict__ evtr, const float* __restrict__ evfix,
                         _Float16* __restrict__ Qh, _Float16* __restrict__ Qeh) {
  size_t idx = (size_t)blockIdx.x * blockDim.x + threadIdx.x;
  if (idx >= (size_t)784 * 784) return;
  int i = (int)(idx / 784), k = (int)(idx % 784);
  float q, e;
  if (k < 774) { q = btr[(size_t)i * 774 + k] * (1.0f / 1.15f); e = evtr[k]; }
  else         { q = bfix[(size_t)i * 10 + (k - 774)];          e = evfix[k - 774]; }
  e = fminf(fmaxf(e, -1000000.0f), 200.0f);
  Qh[idx]  = (_Float16)q;
  Qeh[idx] = (_Float16)(q * e);
}

__global__ void k_scan_init(float* __restrict__ xs, float* __restrict__ ys,
                            _Float16* __restrict__ xh, int n) {
  int i = blockIdx.x * blockDim.x + threadIdx.x;
  if (i >= n) return;
  float x = xs[i];
  ys[i] = x;
  xh[i] = (_Float16)x;
}

__global__ void k_scan_update(const float* __restrict__ z, float* __restrict__ xs,
                              float* __restrict__ ys, _Float16* __restrict__ xh,
                              const float* __restrict__ gamma, int n) {
  int i = blockIdx.x * blockDim.x + threadIdx.x;
  if (i >= n) return;
  const float Dx = 0.0357142857142857f; // 1/sqrt(784)
  float x = xs[i], y = ys[i];
  float I1 = 7.2f * x - 2.0f * y + (-1.2f) + Dx * z[i];
  float I2 = -1.0f * y + 0.1f;
  float fe = 0.25f * tanhf(3.7f * I1) + 0.65f;
  float fi = 0.5f  * tanhf(I2) + 0.5f;
  float xn = x + 0.1f * (-1.5f * x + (1.0f - x) * fe);
  float yn = y + (0.1f / gamma[0]) * (-0.4f * y + (1.0f - y) * fi);
  xs[i] = xn;
  ys[i] = yn;
  xh[i] = (_Float16)xn;
}

// ---------------------------------------------------------------------------
// generic (small-Cin) WMMA conv — used only for layer 0 (Cin=3, K=27)
// ---------------------------------------------------------------------------
__global__ void k_conv_gen(const _Float16* __restrict__ x, const _Float16* __restrict__ wt,
                           const float* __restrict__ bias, _Float16* __restrict__ y,
                           int NB, int H, int W, int Cin, int Cout) {
  const int K9   = 9 * Cin;
  const int lane = threadIdx.x & 31;
  const int wave = threadIdx.x >> 5;
  const int g = lane >> 4, mn = lane & 15;
  const int npix = NB * H * W;
  const int ptile = blockIdx.x * 4 + wave;
  if (ptile * 16 >= npix) return; // wave-uniform

  int p = ptile * 16 + mn;
  if (p >= npix) p = npix - 1;
  const int hw = H * W;
  const int nb = p / hw;
  int rem = p - nb * hw;
  const int oy = rem / W, ox = rem - (rem / W) * W;
  const int ncol = blockIdx.y * 16 + mn;
  const _Float16* wrow = wt + (size_t)ncol * K9;

  v8f acc = {};
  for (int k0 = 0; k0 < K9; k0 += 32) {
    v16h av, bv;
    #pragma unroll
    for (int e = 0; e < 16; e++) {
      int k = k0 + g * 16 + e;
      _Float16 val = (_Float16)0.0f;
      if (k < K9) {
        int r = k / Cin, ci = k - r * Cin;
        int ky = r / 3, kx = r - ky * 3;
        int iy = oy + ky - 1, ix = ox + kx - 1;
        if (iy >= 0 && iy < H && ix >= 0 && ix < W)
          val = x[(((size_t)nb * H + iy) * W + ix) * Cin + ci];
      }
      av[e] = val;
    }
    #pragma unroll
    for (int e = 0; e < 16; e++) {
      int k = k0 + g * 16 + e;
      bv[e] = (k < K9) ? wrow[k] : (_Float16)0.0f;
    }
    acc = WMMA_F16(av, bv, acc);
  }
  float bsum = bias[ncol];
  #pragma unroll
  for (int d = 0; d < 8; d++) {
    int pr = ptile * 16 + g * 8 + d;
    if (pr < npix) {
      float v = acc[d] + bsum;
      y[(size_t)pr * Cout + ncol] = (_Float16)(v > 0.0f ? v : 0.0f);
    }
  }
}

// ---------------------------------------------------------------------------
// main conv (Cin in {64,128,256,512}): block = 8 waves / 256 threads; each
// wave owns a 16-pixel M-tile; all waves share a 64-cout N-block whose 64x32
// f16 weight chunks are double-buffered in LDS via the Tensor Data Mover
// (wave 0 issues D#, s_wait_tensorcnt, workgroup barrier).  A fragment is
// loaded once per k-step and reused across 4 WMMAs.  B fragments are read
// through explicit addrspace(3) volatile pointers (raw byte offsets matching
// the TDM descriptors) so they select to ds_load_b128 and cannot be folded.
// ---------------------------------------------------------------------------
__global__ void k_conv_tdm(const _Float16* __restrict__ x, const _Float16* __restrict__ wt,
                           const float* __restrict__ bias, _Float16* __restrict__ y,
                           int NB, int H, int W, int cinShift, int Cout) {
  __shared__ _Float16 smem[2][64 * 32]; // two 4KB weight chunks at LDS 0 / 4096
  const int Cin = 1 << cinShift;
  const int K9  = 9 * Cin;
  const int lane = threadIdx.x & 31;
  const int wave = threadIdx.x >> 5;
  const int g = lane >> 4, mn = lane & 15;
  const int npix = NB * H * W;
  const int ptile = blockIdx.x * 8 + wave;

  int p = ptile * 16 + mn;
  if (p >= npix) p = npix - 1;          // no early return: barriers below
  const int hw = H * W;
  const int nb = p / hw;
  int rem = p - nb * hw;
  const int oy = rem / W, ox = rem - (rem / W) * W;

  const int ncolBase = blockIdx.y * 64;
  const _Float16* wblk = wt + (size_t)ncolBase * K9;
  const int nsteps = K9 >> 5;           // K9 % 32 == 0 here

  // anchor the LDS allocation (address escapes into asm) so the 8KB stays
  // reserved for the TDM even though normal code never stores to it
  {
    _Float16* esc = &smem[0][0];
    asm volatile("" : "+v"(esc) :: "memory");
  }

#if HAS_TDM
  if (wave == 0)
    tdm_load_2d(0u, wblk, (unsigned)K9, 64u, 32u, 64u, (unsigned)K9);
#endif

  v8f acc0 = {}, acc1 = {}, acc2 = {}, acc3 = {};
  for (int i = 0; i < nsteps; i++) {
    const int k0 = i << 5;
#if HAS_TDM
    if (wave == 0) {
      if (i + 1 < nsteps) {
        tdm_load_2d((unsigned)(((i + 1) & 1) * (64 * 32 * 2)),
                    wblk + (size_t)(i + 1) * 32,
                    (unsigned)K9, 64u, 32u, 64u, (unsigned)K9);
        __builtin_amdgcn_s_wait_tensorcnt(1);
      } else {
        __builtin_amdgcn_s_wait_tensorcnt(0);
      }
    }
    __syncthreads();
    const unsigned ldsBase = (unsigned)(i & 1) * 4096u;
#else
    __syncthreads(); // protect previous chunk reads
    {
      // cooperative copy: threads x 16 halves = 64x32 chunk
      for (int j = threadIdx.x * 16; j < 64 * 32; j += blockDim.x * 16) {
        int r = j >> 5, kk = j & 31;
        *(v8h*)&smem[0][j]     = *(const v8h*)(wblk + (size_t)r * K9 + k0 + kk);
        *(v8h*)&smem[0][j + 8] = *(const v8h*)(wblk + (size_t)r * K9 + k0 + kk + 8);
      }
    }
    __syncthreads();
    const unsigned ldsBase = 0u;
#endif

    // A fragment: two contiguous 8-half runs (channel-inner, Cin%64==0)
    v8h runs[2];
    #pragma unroll
    for (int h = 0; h < 2; h++) {
      int ks = k0 + g * 16 + h * 8;
      int r = ks >> cinShift, ci = ks & (Cin - 1);
      int ky = r / 3, kx = r - ky * 3;
      int iy = oy + ky - 1, ix = ox + kx - 1;
      bool ok = iy >= 0 && iy < H && ix >= 0 && ix < W;
      v8h z = {};
      runs[h] = ok ? *(const v8h*)(x + (((size_t)nb * H + iy) * W + ix) * Cin + ci) : z;
    }
    v16h av = cat16(runs[0], runs[1]);

    // 4 B fragments from LDS: row = cout-in-block (64B apart), 32 halves each
    #pragma unroll
    for (int s = 0; s < 4; s++) {
      unsigned boff = ldsBase + (unsigned)(((s * 16 + mn) << 5) + g * 16) * 2u;
      v8h b0 = lds_read_v8h(boff);
      v8h b1 = lds_read_v8h(boff + 16u);
      v16h bv = cat16(b0, b1);
      if (s == 0) acc0 = WMMA_F16(av, bv, acc0);
      if (s == 1) acc1 = WMMA_F16(av, bv, acc1);
      if (s == 2) acc2 = WMMA_F16(av, bv, acc2);
      if (s == 3) acc3 = WMMA_F16(av, bv, acc3);
    }
    __syncthreads(); // before next chunk overwrites its buffer
  }

  // epilogue: bias + relu + store (NHWC)
  #pragma unroll
  for (int s = 0; s < 4; s++) {
    v8f a = (s == 0) ? acc0 : (s == 1) ? acc1 : (s == 2) ? acc2 : acc3;
    int ncol = ncolBase + s * 16 + mn;
    float bsum = bias[ncol];
    #pragma unroll
    for (int d = 0; d < 8; d++) {
      int pr = ptile * 16 + g * 8 + d;
      if (pr < npix) {
        float v = a[d] + bsum;
        y[(size_t)pr * Cout + ncol] = (_Float16)(v > 0.0f ? v : 0.0f);
      }
    }
  }
}

// ---------------------------------------------------------------------------
// dense: y[8xN] = relu(x[8xK] @ W[KxN] + b).  Single-use f32 weights -> direct
// strided loads (coalesced across the 16 lanes of each column group).
// ---------------------------------------------------------------------------
__global__ void k_dense(const _Float16* __restrict__ x, const float* __restrict__ w,
                        const float* __restrict__ bias, _Float16* __restrict__ y,
                        float* __restrict__ y32, int K, int N) {
  const int lane = threadIdx.x & 31;
  const int wave = threadIdx.x >> 5;
  const int g = lane >> 4, mn = lane & 15;
  const int ntiles = (N + 15) / 16;
  const int ntile = blockIdx.x * 4 + wave;
  if (ntile >= ntiles) return; // wave-uniform
  const int n = ntile * 16 + mn;
  const bool nok = n < N;
  const _Float16* xrow = x + (size_t)(mn & 7) * K; // rows 8..15 mirror 0..7 (unstored)

  v8f acc = {};
  for (int k0 = 0; k0 < K; k0 += 32) {
    v16h av = mkA(xrow, k0, g, K);
    v16h bv;
    #pragma unroll
    for (int e = 0; e < 16; e++) {
      int k = k0 + g * 16 + e;
      bv[e] = (nok && k < K) ? (_Float16)w[(size_t)k * N + n] : (_Float16)0.0f;
    }
    acc = WMMA_F16(av, bv, acc);
  }
  #pragma unroll
  for (int d = 0; d < 8; d++) {
    int M = g * 8 + d;
    if (M < 8 && nok) {
      float v = acc[d] + bias[n];
      if (v < 0.0f) v = 0.0f;
      y[(size_t)M * N + n] = (_Float16)v;
      if (y32) y32[(size_t)M * N + n] = v;
    }
  }
}

// C[MxN] = A[MxK] * Bt[NxK]^T (f16 in/out) — matA = Qe @ Q^T; M,N % 16 == 0
__global__ void k_gemm_tt(const _Float16* __restrict__ A, const _Float16* __restrict__ Bt,
                          _Float16* __restrict__ C, int M, int N, int K) {
  const int lane = threadIdx.x & 31;
  const int wave = threadIdx.x >> 5;
  const int g = lane >> 4, mn = lane & 15;
  const int ntile = blockIdx.y * 4 + wave;
  if (ntile * 16 >= N) return; // wave-uniform
  const int m = blockIdx.x * 16 + mn;
  const int n = ntile * 16 + mn;
  const _Float16* arow = A + (size_t)m * K;
  const _Float16* brow = Bt + (size_t)n * K;

  v8f acc = {};
  for (int k0 = 0; k0 < K; k0 += 32) {
    acc = WMMA_F16(mkA(arow, k0, g, K), mkB(brow, k0, g, K), acc);
  }
  #pragma unroll
  for (int d = 0; d < 8; d++) {
    int Mr = blockIdx.x * 16 + g * 8 + d;
    C[(size_t)Mr * N + n] = (_Float16)acc[d];
  }
}

// z[8x784] = xh[8x784] @ matA (symmetric, row-major f16)
__global__ void k_sgemm(const _Float16* __restrict__ xh, const _Float16* __restrict__ Bt,
                        float* __restrict__ z, int K, int N) {
  const int lane = threadIdx.x & 31;
  const int wave = threadIdx.x >> 5;
  const int g = lane >> 4, mn = lane & 15;
  const int ntile = blockIdx.x * 4 + wave;
  if (ntile * 16 >= N) return; // wave-uniform
  const int n = ntile * 16 + mn;
  const _Float16* arow = xh + (size_t)(mn & 7) * K;
  const _Float16* brow = Bt + (size_t)n * K;

  v8f acc = {};
  for (int k0 = 0; k0 < K; k0 += 32) {
    acc = WMMA_F16(mkA(arow, k0, g, K), mkB(brow, k0, g, K), acc);
  }
  #pragma unroll
  for (int d = 0; d < 8; d++) {
    int M = g * 8 + d;
    if (M < 8) z[(size_t)M * N + n] = acc[d];
  }
}

// ---------------------------------------------------------------------------
extern "C" void kernel_launch(void* const* d_in, const int* in_sizes, int n_in,
                              void* d_out, int out_size, void* d_ws, size_t ws_size,
                              hipStream_t stream) {
  // dict flattening: 0 inputs, 1..13 conv_ks, 14..26 conv_bs,
  // 27..42 bn (g,b,m,v)x4, 43..46 dense_Ws, 47..50 dense_bs,
  // 51 base_train, 52 base_fix, 53 autov_tr, 54 autov_fix, 55 gamma
  const float* inp = (const float*)d_in[0];
  const float* convK[13]; const float* convBia[13];
  for (int i = 0; i < 13; i++) { convK[i] = (const float*)d_in[1 + i]; convBia[i] = (const float*)d_in[14 + i]; }
  const float *bnG[4], *bnB[4], *bnM[4], *bnV[4];
  for (int j = 0; j < 4; j++) {
    bnG[j] = (const float*)d_in[27 + 4 * j + 0];
    bnB[j] = (const float*)d_in[27 + 4 * j + 1];
    bnM[j] = (const float*)d_in[27 + 4 * j + 2];
    bnV[j] = (const float*)d_in[27 + 4 * j + 3];
  }
  const float* dW[4]; const float* dB[4];
  for (int j = 0; j < 4; j++) { dW[j] = (const float*)d_in[43 + j]; dB[j] = (const float*)d_in[47 + j]; }
  const float* btr   = (const float*)d_in[51];
  const float* bfix  = (const float*)d_in[52];
  const float* evtr  = (const float*)d_in[53];
  const float* evfix = (const float*)d_in[54];
  const float* gam   = (const float*)d_in[55];

  char* base = (char*)d_ws;
  size_t off = 0;
  auto alloc = [&](size_t bytes) -> void* {
    void* p = base + off;
    off = (off + bytes + 255) & ~(size_t)255;
    return p;
  };
  const size_t actElems = (size_t)8 * 224 * 224 * 64;
  _Float16* bufA  = (_Float16*)alloc(actElems * 2);
  _Float16* bufB  = (_Float16*)alloc(actElems * 2);
  _Float16* wtC   = (_Float16*)alloc((size_t)9 * 512 * 512 * 2);
  _Float16* xbn   = (_Float16*)alloc((size_t)8 * 25088 * 2);
  _Float16* actD  = (_Float16*)alloc((size_t)8 * 4096 * 2);
  _Float16* Qh    = (_Float16*)alloc((size_t)784 * 784 * 2);
  _Float16* Qeh   = (_Float16*)alloc((size_t)784 * 784 * 2);
  _Float16* matAh = (_Float16*)alloc((size_t)784 * 784 * 2);
  float*    xs    = (float*)alloc((size_t)6272 * 4);
  float*    ys    = (float*)alloc((size_t)6272 * 4);
  float*    z     = (float*)alloc((size_t)6272 * 4);
  _Float16* xh    = (_Float16*)alloc((size_t)6272 * 2);

  // 1) input f32 -> f16
  size_t nin = (size_t)8 * 224 * 224 * 3;
  k_f32_to_f16<<<(unsigned)((nin + 255) / 256), 256, 0, stream>>>(inp, bufA, nin);

  // 2) VGG conv stack
  const int cc[13]  = {64, 64, 128, 128, 256, 256, 256, 512, 512, 512, 512, 512, 512};
  const int cin[13] = {3, 64, 64, 128, 128, 256, 256, 256, 512, 512, 512, 512, 512};
  const bool pool[13] = {false, true, false, true, false, false, true,
                         false, false, true, false, false, true};
  _Float16* cur = bufA; _Float16* nxt = bufB;
  int H = 224;
  for (int l = 0; l < 13; l++) {
    size_t wn = (size_t)9 * cin[l] * cc[l];
    k_wconv_t<<<(unsigned)((wn + 255) / 256), 256, 0, stream>>>(convK[l], wtC, cin[l], cc[l]);
    int npix = 8 * H * H;
    int ptiles = npix / 16;
    if (l == 0) {
      dim3 gr((unsigned)((ptiles + 3) / 4), (unsigned)(cc[l] / 16));
      k_conv_gen<<<gr, 128, 0, stream>>>(cur, wtC, convBia[l], nxt, 8, H, H, cin[l], cc[l]);
    } else {
      int cinShift = 31 - __builtin_clz((unsigned)cin[l]);
      dim3 gr((unsigned)((ptiles + 7) / 8), (unsigned)(cc[l] / 64));
      k_conv_tdm<<<gr, 256, 0, stream>>>(cur, wtC, convBia[l], nxt, 8, H, H, cinShift, cc[l]);
    }
    _Float16* t = cur; cur = nxt; nxt = t;
    if (pool[l]) {
      int OH = H / 2;
      size_t n = (size_t)8 * OH * OH * cc[l];
      k_maxpool<<<(unsigned)((n + 255) / 256), 256, 0, stream>>>(cur, nxt, 8, H, H, cc[l]);
      t = cur; cur = nxt; nxt = t;
      H = OH;
    }
  }

  // 3) BN + dense layers
  const int dims[5] = {25088, 4096, 2024, 1048, 784};
  _Float16* dcur = cur;
  for (int j = 0; j < 4; j++) {
    int K = dims[j], N = dims[j + 1];
    size_t nb = (size_t)8 * K;
    k_bn<<<(unsigned)((nb + 255) / 256), 256, 0, stream>>>(dcur, bnG[j], bnB[j], bnM[j], bnV[j], xbn, K);
    int ntiles = (N + 15) / 16;
    k_dense<<<(unsigned)((ntiles + 3) / 4), 128, 0, stream>>>(
        xbn, dW[j], dB[j], actD, (j == 3) ? xs : (float*)nullptr, K, N);
    dcur = actD;
  }

  // 4) matA = Q diag(eig) Q^T (analytic inverse of base = Q diag(s))
  size_t nq = (size_t)784 * 784;
  k_buildQ<<<(unsigned)((nq + 255) / 256), 256, 0, stream>>>(btr, bfix, evtr, evfix, Qh, Qeh);
  dim3 gq(49, 13);
  k_gemm_tt<<<gq, 128, 0, stream>>>(Qeh, Qh, matAh, 784, 784, 784);

  // 5) 100-step Wilson-Cowan scan
  k_scan_init<<<(6272 + 255) / 256, 256, 0, stream>>>(xs, ys, xh, 6272);
  for (int t = 0; t < 100; t++) {
    k_sgemm<<<13, 128, 0, stream>>>(xh, matAh, z, 784, 784);
    k_scan_update<<<(6272 + 255) / 256, 256, 0, stream>>>(z, xs, ys, xh, gam, 6272);
  }

  // 6) output (f32, 8x784)
  hipMemcpyAsync(d_out, xs, (size_t)6272 * sizeof(float), hipMemcpyDeviceToDevice, stream);
}
